// WOQ_Linear_Unary_Model_82841329206003
// MI455X (gfx1250) — compile-verified, matrix operand order checked by reading, not run on hardware
//
#include <hip/hip_runtime.h>

typedef __attribute__((ext_vector_type(16))) _Float16 v16h;
typedef __attribute__((ext_vector_type(8)))  _Float16 v8h;
typedef __attribute__((ext_vector_type(8)))  float    v8f;

#define D_IN   4096
#define D_OUT  16384
#define M_TOT  8192          // B*S = 4*2048

#define BM 128               // block M tile (8 fragments per wave)
#define BN 128               // block N tile (8 waves * 16)
#define BK 64                // K step per iteration = 2 WMMA K-slices
#define MFRAGS 8             // BM / 16
#define LDS_STRIDE 72        // halves per row (144B = 9*16B, 36 dwords -> bank spread)

// Branch-free erf (A&S 7.1.26, max abs err 1.5e-7) -> exact-erf GELU.
// Emitted once (noinline); called per 16x16 fragment.
__device__ __attribute__((noinline))
void gelu_store8(v8f c, float s, float* __restrict__ p)
{
    #pragma unroll
    for (int r = 0; r < 8; ++r) {
        float v  = c[r] * s;
        float z  = v * 0.70710678118654752440f;
        float az = __builtin_fabsf(z);
        float t  = __builtin_amdgcn_rcpf(__builtin_fmaf(0.3275911f, az, 1.0f));
        float poly = __builtin_fmaf(1.061405429f, t, -1.453152027f);
        poly = __builtin_fmaf(poly, t,  1.421413741f);
        poly = __builtin_fmaf(poly, t, -0.284496736f);
        poly = __builtin_fmaf(poly, t,  0.254829592f);
        poly = poly * t;
        float e    = __expf(-z * z);
        float erfv = __builtin_copysignf(__builtin_fmaf(-poly, e, 1.0f), z);
        float g    = 0.5f * v * (1.0f + erfv);
        __builtin_nontemporal_store(g, p + (size_t)r * D_OUT);
    }
}

// 16 int32 codes (K: kb..kb+7, kb+16..kb+23) -> f16 (q-8), exact, via the
// f16 magic-number trick: (q | 0x6400) as f16 == 1024+q for q in [0,1024);
// packed subtract of 1032.0 (exact in f16) yields q-8. 3 VALU per pair.
__device__ __forceinline__ v16h dequant16(const int* __restrict__ qp)
{
    int4 q0 = *(const int4*)(qp);
    int4 q1 = *(const int4*)(qp + 4);
    int4 q2 = *(const int4*)(qp + 16);
    int4 q3 = *(const int4*)(qp + 20);
    union { unsigned u[8]; v16h v; } r;
    const unsigned M = 0x64006400u;   // f16 1024.0 | 1024.0
    r.u[0] = ((unsigned)q0.x | ((unsigned)q0.y << 16)) | M;
    r.u[1] = ((unsigned)q0.z | ((unsigned)q0.w << 16)) | M;
    r.u[2] = ((unsigned)q1.x | ((unsigned)q1.y << 16)) | M;
    r.u[3] = ((unsigned)q1.z | ((unsigned)q1.w << 16)) | M;
    r.u[4] = ((unsigned)q2.x | ((unsigned)q2.y << 16)) | M;
    r.u[5] = ((unsigned)q2.z | ((unsigned)q2.w << 16)) | M;
    r.u[6] = ((unsigned)q3.x | ((unsigned)q3.y << 16)) | M;
    r.u[7] = ((unsigned)q3.z | ((unsigned)q3.w << 16)) | M;
    v16h bias;
    #pragma unroll
    for (int i = 0; i < 16; ++i) bias[i] = (_Float16)1032.0f;
    return r.v - bias;                 // 8x v_pk_add_f16
}

__global__ __launch_bounds__(256)
void woq_gemm_gelu(const float* __restrict__ x,
                   const int*   __restrict__ qw,
                   const float* __restrict__ scales,
                   float*       __restrict__ out)
{
    __shared__ __align__(16) _Float16 aTile[BM * LDS_STRIDE];

    const int tid    = threadIdx.x;
    const int wave   = tid >> 5;
    const int lane   = tid & 31;
    const int nlocal = lane & 15;     // N within 16-wide strip (B/C/D layouts)
    const int hi     = lane >> 4;     // 0: K 0-7/16-23, M rows 0-7 ; 1: K 8-15/24-31, rows 8-15
    const int kb     = hi * 8;

    const int m0   = blockIdx.y * BM;
    const int n0   = blockIdx.x * BN;
    const int ncol = n0 + wave * 16 + nlocal;   // this lane's output channel

    // cooperative A-stage mapping: 128 rows x 64 k, 32 f32->f16 per thread
    const int arow  = tid >> 1;
    const int akseg = (tid & 1) * 32;

    const int*   qbase = qw + (size_t)ncol * D_IN;
    const float* xrow  = x  + (size_t)(m0 + arow) * D_IN;

    v8f acc[MFRAGS] = {};   // 8 stacked 16x16 M-fragments, f32 accum

    for (int k0 = 0; k0 < D_IN; k0 += BK) {
        __syncthreads();
        // ---- stage x tile (128x64) into LDS as f16 ----
        {
            const float* xp = xrow + k0 + akseg;
            _Float16*    dp = aTile + arow * LDS_STRIDE + akseg;
            #pragma unroll
            for (int c = 0; c < 4; ++c) {
                float4 f0 = *(const float4*)(xp + c * 8);
                float4 f1 = *(const float4*)(xp + c * 8 + 4);
                v8h h;
                h[0] = (_Float16)f0.x; h[1] = (_Float16)f0.y;
                h[2] = (_Float16)f0.z; h[3] = (_Float16)f0.w;
                h[4] = (_Float16)f1.x; h[5] = (_Float16)f1.y;
                h[6] = (_Float16)f1.z; h[7] = (_Float16)f1.w;
                *(v8h*)(dp + c * 8) = h;
            }
        }

        // prefetch next qweight K-slab while staging
        if (k0 + BK < D_IN)
            __builtin_prefetch(qbase + k0 + BK + kb, 0, 1);

        // ---- dequantize two B fragments (K slices k0, k0+32) ----
        v16h b0 = dequant16(qbase + k0 + kb);
        v16h b1 = dequant16(qbase + k0 + 32 + kb);

        __syncthreads();

        // ---- 16 WMMAs: 8 M-fragments x 2 K-slices ----
        #pragma unroll
        for (int t = 0; t < MFRAGS; ++t) {
            const _Float16* ap = aTile + (t * 16 + nlocal) * LDS_STRIDE + kb;
            union { v16h v; v8h h[2]; } u;
            u.h[0] = *(const v8h*)(ap);        // K kb..kb+7
            u.h[1] = *(const v8h*)(ap + 16);   // K kb+16..kb+23
            acc[t] = __builtin_amdgcn_wmma_f32_16x16x32_f16(
                         false, u.v, false, b0, (short)0, acc[t], false, false);
        }
        #pragma unroll
        for (int t = 0; t < MFRAGS; ++t) {
            const _Float16* ap = aTile + (t * 16 + nlocal) * LDS_STRIDE + 32 + kb;
            union { v16h v; v8h h[2]; } u;
            u.h[0] = *(const v8h*)(ap);
            u.h[1] = *(const v8h*)(ap + 16);
            acc[t] = __builtin_amdgcn_wmma_f32_16x16x32_f16(
                         false, u.v, false, b1, (short)0, acc[t], false, false);
        }
    }

    // ---- epilogue: per-channel scale + exact-erf GELU + NT stores ----
    const float s = scales[ncol];
    #pragma unroll
    for (int t = 0; t < MFRAGS; ++t) {
        const int mbase = m0 + t * 16 + hi * 8;   // C/D layout: VGPR r -> row r (+8 upper half)
        gelu_store8(acc[t], s, out + (size_t)mbase * D_OUT + ncol);
    }
}

extern "C" void kernel_launch(void* const* d_in, const int* in_sizes, int n_in,
                              void* d_out, int out_size, void* d_ws, size_t ws_size,
                              hipStream_t stream) {
    const float* x      = (const float*)d_in[0];
    const int*   qw     = (const int*)  d_in[1];
    const float* scales = (const float*)d_in[2];
    float*       out    = (float*)      d_out;

    dim3 grid(D_OUT / BN, M_TOT / BM);   // (128, 64)
    woq_gemm_gelu<<<grid, dim3(256), 0, stream>>>(x, qw, scales, out);
}
